// CausalAttention_41016937677090
// MI455X (gfx1250) — matrix operand order repaired and checked by reading, not compiled, not run on hardware
//
#include <hip/hip_runtime.h>

// CDNA5 / gfx1250, wave32. Causal attention B=4,S=4096,D=1024,H=128.
// bf16 WMMA (v_wmma_f32_16x16x32_bf16) for all three GEMM stages,
// f32 flash-softmax with 64-key tiles to amortize VALU overhead.

typedef __bf16 bf16;
typedef __attribute__((ext_vector_type(16))) __bf16 bf16x16;
typedef __attribute__((ext_vector_type(8)))  __bf16 bf16x8;
typedef __attribute__((ext_vector_type(8)))  float  f32x8;

#define SEQ 4096
#define EMB 1024
#define HS  128
#define NB  4

__device__ __forceinline__ bf16 f2bf(float f) {
    unsigned u = __builtin_bit_cast(unsigned, f);
    u += 0x7FFFu + ((u >> 16) & 1u);        // round-to-nearest-even
    unsigned short h = (unsigned short)(u >> 16);
    return __builtin_bit_cast(bf16, h);
}

__device__ __forceinline__ f32x8 wmma_bf16(bf16x16 a, bf16x16 b, f32x8 c) {
    // (neg_a, A, neg_b, B, c_mod, C, reuse_a, reuse_b)
    return __builtin_amdgcn_wmma_f32_16x16x32_bf16(false, a, false, b,
                                                   (short)0, c, false, false);
}

// pack two contiguous 8 x bf16 (16B) chunks into a 16-element fragment
__device__ __forceinline__ bf16x16 pack16(const bf16* lo, const bf16* hi) {
    bf16x8 a0 = *(const bf16x8*)lo;
    bf16x8 a1 = *(const bf16x8*)hi;
    bf16x16 r;
#pragma unroll
    for (int i = 0; i < 8; ++i) { r[i] = a0[i]; r[i + 8] = a1[i]; }
    return r;
}

// ---------------- stage 0: f32 -> bf16 conversion of x ----------------
__global__ __launch_bounds__(256) void cvt_x(const float* __restrict__ x,
                                             bf16* __restrict__ xb, long n) {
    long i = (long)blockIdx.x * 256 + threadIdx.x;
    if (i < n) xb[i] = f2bf(x[i]);
}

// ------- stage 0b: weights f32 [D,H] -> bf16 transposed [H,D] ---------
__global__ __launch_bounds__(256) void cvt_w(const float* __restrict__ Wq,
                                             const float* __restrict__ Wk,
                                             const float* __restrict__ Wv,
                                             bf16* __restrict__ WqT,
                                             bf16* __restrict__ WkT,
                                             bf16* __restrict__ WvT) {
    int idx = blockIdx.x * 256 + threadIdx.x;   // over D*H = 131072
    int d = idx >> 7;
    int h = idx & (HS - 1);
    int t = h * EMB + d;
    WqT[t] = f2bf(Wq[idx]);
    WkT[t] = f2bf(Wk[idx]);
    WvT[t] = f2bf(Wv[idx]);
}

// ---------------- stage 1: QKV projection (bf16 WMMA) ----------------
// grid (M/16, 1, 3), block 256 = 8 waves, wave w owns N-tile w (H=128).
__global__ __launch_bounds__(256) void proj_qkv(
        const bf16* __restrict__ xb,                      // [B*S, D]
        const bf16* __restrict__ WqT, const bf16* __restrict__ WkT,
        const bf16* __restrict__ WvT,
        const float* __restrict__ bq, const float* __restrict__ bk,
        const float* __restrict__ bv,
        bf16* __restrict__ Qb,                            // [B*S, H]
        bf16* __restrict__ Kb,                            // [B*S, H]
        bf16* __restrict__ VTb)                           // [B, H, S]
{
    const int mt   = blockIdx.x;
    const int z    = blockIdx.z;              // 0=Q 1=K 2=V
    const int wave = threadIdx.x >> 5;        // N-tile
    const int lane = threadIdx.x & 31;
    const int half = lane >> 4;
    const int l15  = lane & 15;

    const bf16*  WT   = (z == 0) ? WqT : ((z == 1) ? WkT : WvT);
    const float* bias = (z == 0) ? bq  : ((z == 1) ? bk  : bv);

    const int m = mt * 16 + l15;              // A row (both lane halves hold M=0..15)
    const int n = wave * 16 + l15;            // B/C column

    f32x8 acc = {};
    const bf16* arow = xb + (long)m * EMB;
    const bf16* brow = WT + (long)n * EMB;
#pragma unroll 4
    for (int k0 = 0; k0 < EMB; k0 += 32) {
        bf16x16 A = pack16(arow + k0 + half * 8, arow + k0 + half * 8 + 16);
        bf16x16 B = pack16(brow + k0 + half * 8, brow + k0 + half * 8 + 16);
        acc = wmma_bf16(A, B, acc);
    }

    const float bn = bias[n];
    if (z < 2) {
        bf16* outp = (z == 0) ? Qb : Kb;
#pragma unroll
        for (int r = 0; r < 8; ++r) {
            int row = mt * 16 + r + 8 * half;
            outp[(long)row * HS + n] = f2bf(acc[r] + bn);
        }
    } else {
#pragma unroll
        for (int r = 0; r < 8; ++r) {
            int s  = mt * 16 + r + 8 * half;
            int bb = s >> 12;                 // s / SEQ
            int si = s & (SEQ - 1);
            VTb[((long)bb * HS + n) * SEQ + si] = f2bf(acc[r] + bn);
        }
    }
}

// ---------------- stage 2: flash attention (bf16 WMMA) ----------------
// grid (S/16, B), block 32 (one wave per 16-query tile), 64-key tiles.
__global__ __launch_bounds__(32) void attn(
        const bf16* __restrict__ Qb,          // [B*S, H]
        const bf16* __restrict__ Kb,          // [B*S, H]
        const bf16* __restrict__ VTb,         // [B, H, S]
        float* __restrict__ out)              // [1,B,S,H] == [B,S,H]
{
    __shared__ __align__(16) bf16 lds_p[16 * 64];   // 16x64 P tile (2 KB)

    const int lane  = threadIdx.x & 31;
    const int half  = lane >> 4;
    const int l15   = lane & 15;
    const int qt    = blockIdx.x;
    const int b     = blockIdx.y;
    const int qbase = qt * 16;

    // Q A-fragments: 4 K-steps over head dim 128
    const bf16* qrow = Qb + (long)(b * SEQ + qbase + l15) * HS;
    bf16x16 Afrag[4];
#pragma unroll
    for (int ks = 0; ks < 4; ++ks) {
        const bf16* p = qrow + ks * 32 + half * 8;
        Afrag[ks] = pack16(p, p + 16);
    }

    f32x8 Oacc[8] = {};
    float mrow[8], lrow[8];
#pragma unroll
    for (int r = 0; r < 8; ++r) { mrow[r] = -3.0e38f; lrow[r] = 0.0f; }

    const float sinv = 0.08838834764831845f;  // 1/sqrt(128)
    const int ntiles = (qbase + 79) >> 6;     // 64-key tiles, causal bound

    const bf16* kbat = Kb + (long)b * SEQ * HS;
    const bf16* vcol = VTb + ((long)b * HS + l15) * SEQ + half * 16;

    for (int kt = 0; kt < ntiles; ++kt) {
        const int kbase = kt << 6;

        // prefetch next tile's K rows / V columns while we compute
        if (kt + 1 < ntiles) {
            __builtin_prefetch((const void*)(kbat + (long)(kbase + 64 + l15) * HS), 0, 0);
            __builtin_prefetch((const void*)(kbat + (long)(kbase + 96 + l15) * HS), 0, 0);
            __builtin_prefetch((const void*)(vcol + kbase + 64), 0, 0);
        }

        // ---- S = Q K^T : four 16x16 tiles (keys kbase+16*sub) ----
        f32x8 S[4] = {};
#pragma unroll
        for (int ks = 0; ks < 4; ++ks) {
#pragma unroll
            for (int sub = 0; sub < 4; ++sub) {
                const bf16* kr = kbat + (long)(kbase + sub * 16 + l15) * HS
                                      + ks * 32 + half * 8;
                bf16x16 B = pack16(kr, kr + 16);
                S[sub] = wmma_bf16(Afrag[ks], B, S[sub]);
            }
        }

        // ---- scale + causal mask + online softmax (once per 64 keys) ----
#pragma unroll
        for (int r = 0; r < 8; ++r) {
            const int qg = qbase + r + 8 * half;   // row of this element
            float sv[4];
#pragma unroll
            for (int sub = 0; sub < 4; ++sub) {
                const int kg = kbase + sub * 16 + l15;
                sv[sub] = (kg <= qg) ? S[sub][r] * sinv : -3.0e38f;
            }
            float mx = fmaxf(fmaxf(sv[0], sv[1]), fmaxf(sv[2], sv[3]));
#pragma unroll
            for (int off = 1; off < 16; off <<= 1)
                mx = fmaxf(mx, __shfl_xor(mx, off, 32));
            const float mnew = fmaxf(mrow[r], mx);
            const float corr = __expf(mrow[r] - mnew);
            const int ro = (r + 8 * half) * 64 + l15;
            float ps = 0.0f;
#pragma unroll
            for (int sub = 0; sub < 4; ++sub) {
                float p = __expf(sv[sub] - mnew);
                ps += p;
                lds_p[ro + sub * 16] = (bf16)p;    // direct C->row-major store
            }
#pragma unroll
            for (int off = 1; off < 16; off <<= 1)
                ps += __shfl_xor(ps, off, 32);
            lrow[r] = lrow[r] * corr + ps;
            mrow[r] = mnew;
#pragma unroll
            for (int ht = 0; ht < 8; ++ht) Oacc[ht][r] *= corr;
        }

        __syncthreads();   // single-wave WG: cheap; orders LDS store->load

        // ---- P A-fragments (two 16x32 chunks of the 16x64 tile) ----
        const bf16* pp0 = lds_p + l15 * 64 + half * 16;
        bf16x16 Pf0 = pack16(pp0, pp0 + 8);
        bf16x16 Pf1 = pack16(pp0 + 32, pp0 + 40);

        // ---- O += P V  (K-dim = 64 keys, 8 head tiles) ----
        const bf16* vk = vcol + kbase;
#pragma unroll
        for (int ht = 0; ht < 8; ++ht) {
            const bf16* vp = vk + (long)(ht * 16) * SEQ;
            bf16x16 Bv0 = pack16(vp, vp + 8);
            bf16x16 Bv1 = pack16(vp + 32, vp + 40);
            Oacc[ht] = wmma_bf16(Pf0, Bv0, Oacc[ht]);
            Oacc[ht] = wmma_bf16(Pf1, Bv1, Oacc[ht]);
        }
        __syncthreads();   // orders LDS load -> next iteration's store
    }

    // ---- normalize: softmax/l plus the reference's extra 1/sqrt(H) ----
    float inv[8];
#pragma unroll
    for (int r = 0; r < 8; ++r)
        inv[r] = 1.0f / (lrow[r] * 11.313708498984761f);   // l * sqrt(128)

    const long obase = ((long)b * SEQ + qbase) * HS;
#pragma unroll
    for (int ht = 0; ht < 8; ++ht)
#pragma unroll
        for (int r = 0; r < 8; ++r)
            out[obase + (long)(r + 8 * half) * HS + ht * 16 + l15] =
                Oacc[ht][r] * inv[r];
}

// ---------------------------- launcher ----------------------------
extern "C" void kernel_launch(void* const* d_in, const int* in_sizes, int n_in,
                              void* d_out, int out_size, void* d_ws, size_t ws_size,
                              hipStream_t stream) {
    (void)in_sizes; (void)n_in; (void)out_size; (void)ws_size;
    const float* x  = (const float*)d_in[0];
    const float* Wq = (const float*)d_in[1];
    const float* bq = (const float*)d_in[2];
    const float* Wk = (const float*)d_in[3];
    const float* bk = (const float*)d_in[4];
    const float* Wv = (const float*)d_in[5];
    const float* bv = (const float*)d_in[6];
    float* out = (float*)d_out;

    char* ws = (char*)d_ws;
    bf16* xb  = (bf16*)(ws + 0);                    // 33,554,432 B
    bf16* WqT = (bf16*)(ws + 33554432);             //    262,144 B
    bf16* WkT = (bf16*)(ws + 33816576);             //    262,144 B
    bf16* WvT = (bf16*)(ws + 34078720);             //    262,144 B
    bf16* Qb  = (bf16*)(ws + 34340864);             //  4,194,304 B
    bf16* Kb  = (bf16*)(ws + 38535168);             //  4,194,304 B
    bf16* VTb = (bf16*)(ws + 42729472);             //  4,194,304 B  (total ~46.9 MB)

    const long nx = (long)NB * SEQ * EMB;           // 16,777,216
    cvt_x<<<dim3((unsigned)(nx / 256)), dim3(256), 0, stream>>>(x, xb, nx);
    cvt_w<<<dim3((EMB * HS) / 256), dim3(256), 0, stream>>>(Wq, Wk, Wv, WqT, WkT, WvT);
    proj_qkv<<<dim3((NB * SEQ) / 16, 1, 3), dim3(256), 0, stream>>>(
        xb, WqT, WkT, WvT, bq, bk, bv, Qb, Kb, VTb);
    attn<<<dim3(SEQ / 16, NB), dim3(32), 0, stream>>>(Qb, Kb, VTb, out);
}
